// DecomposedShiftNet_32512902431215
// MI455X (gfx1250) — compile-verified, matrix operand order checked by reading, not run on hardware
//
#include <hip/hip_runtime.h>
#include <math.h>

// ---------------------------------------------------------------------------
// DecomposedShiftNet fused for MI455X (gfx1250, wave32, WMMA f16->f32).
// Weights pre-swizzled once into the per-lane WMMA fragment layout; per-chunk
// LDS staging is a raw async global->LDS copy, double-buffered so the L2->LDS
// stream hides under the previous chunk's WMMAs. Fragment loads are
// contiguous ds_load_b128 pairs.
// ---------------------------------------------------------------------------

typedef _Float16 half16  __attribute__((ext_vector_type(16)));
typedef _Float16 half8   __attribute__((ext_vector_type(8)));
typedef _Float16 half4   __attribute__((ext_vector_type(4)));
typedef float    floatx8 __attribute__((ext_vector_type(8)));
typedef float    floatx4 __attribute__((ext_vector_type(4)));
typedef int      intx4   __attribute__((ext_vector_type(4)));

// Address-space qualified int4 for the async global->LDS builtin.
typedef __attribute__((address_space(1))) intx4 as1_intx4;
typedef __attribute__((address_space(3))) intx4 as3_intx4;

#define BATCH 2048

#if __has_builtin(__builtin_amdgcn_global_load_async_to_lds_b128)
#define HAVE_ASYNC_LDS 1
#else
#define HAVE_ASYNC_LDS 0
#endif

__device__ inline void wait_asynccnt0() {
#if __has_builtin(__builtin_amdgcn_s_wait_asynccnt)
    __builtin_amdgcn_s_wait_asynccnt(0);
#else
    asm volatile("s_wait_asynccnt 0" ::: "memory");
#endif
}

__device__ inline floatx8 wmma_f16(half16 a, half16 b, floatx8 c) {
    return __builtin_amdgcn_wmma_f32_16x16x32_f16(
        false, a, false, b, (short)0, c, false, false);
}

// A fragment: 16x32 f16 tile. ISA 7.12.2: lane l -> m=l%16, g=l/16; the 16
// halfs of lane l are the two contiguous K-runs [kbase+8g, +8) and
// [kbase+16+8g, +8) of row m -> two 128-bit LDS loads.
__device__ inline half16 load_a_frag(const _Float16* __restrict__ lds,
                                     int row_base, int ld, int kbase) {
    const int lane = threadIdx.x & 31;
    const int m = row_base + (lane & 15);
    const int g = lane >> 4;
    const _Float16* p = lds + (size_t)m * ld + kbase + 8 * g;
    half8 lo = *(const half8*)(p);
    half8 hi = *(const half8*)(p + 16);
    return __builtin_shufflevector(lo, hi, 0, 1, 2, 3, 4, 5, 6, 7,
                                   8, 9, 10, 11, 12, 13, 14, 15);
}

// B fragment from fragment-major swizzled staging: chunk layout is
// [ntile][lane][16 halfs] -> one contiguous 32B read per lane.
__device__ inline half16 load_b_frag(const _Float16* __restrict__ wstage,
                                     int ntile) {
    const int lane = threadIdx.x & 31;
    return *(const half16*)(wstage + (size_t)ntile * 512 + lane * 16);
}

// ---------------------------------------------------------------------------
// One-off weight swizzle: f32 (K x N row-major) -> f16 fragment-major chunks.
// Element (k, n): chunk kc=k/32; within chunk kl=k%32: g=(kl>>3)&1,
// kk=kl-8g, j = kk<8 ? kk/2 : 4+(kk-16)/2, h=kl&1, lane=(n%16)+16g.
// dst = kc*32*N + (n/16)*512 + lane*16 + j*2 + h.
// ---------------------------------------------------------------------------
__global__ __launch_bounds__(256) void k_swizzle(const float* __restrict__ W,
                                                 _Float16* __restrict__ out,
                                                 int K, int N) {
    const int e = blockIdx.x * 256 + threadIdx.x;
    if (e >= K * N) return;
    const int k = e / N, n = e % N;
    const int kc = k >> 5, kl = k & 31;
    const int g = (kl >> 3) & 1;
    const int kk = kl - 8 * g;
    const int j = (kk < 8) ? (kk >> 1) : (4 + ((kk - 16) >> 1));
    const int h = kl & 1;
    const size_t dst = (size_t)kc * 32 * N + (size_t)(n >> 4) * 512 +
                       (size_t)((n & 15) + 16 * g) * 16 + j * 2 + h;
    out[dst] = (_Float16)W[e];
}

// ---------------------------------------------------------------------------
// 64 x N = (64 x KTOT) @ (KTOT x N); Wh is pre-swizzled f16; K streamed
// through LDS in 32-row chunks (double-buffered async global->LDS).
// 8 waves: wave w owns M-tile (w&3) and N/32 N-tiles from (w>>2)*(N/32).
// ---------------------------------------------------------------------------
template <int N, int KTOT>
__device__ void gemm64(const _Float16* __restrict__ src, int src_ld,
                       const _Float16* __restrict__ Wh,   // swizzled KTOT x N
                       const float* __restrict__ bias,    // N or nullptr
                       _Float16* __restrict__ ws0,        // >= 32*N halfs
                       _Float16* __restrict__ ws1,        // >= 32*N halfs
                       _Float16* __restrict__ dstL,       // 64 x N f16 LDS
                       float* __restrict__ dstG, int gld, // f32 global
                       bool relu) {
    constexpr int NT = N / 32;
    constexpr int KC = KTOT / 32;
    const int tid = threadIdx.x;
    const int wid = tid >> 5;
    const int lane = tid & 31;
    const int mtile = wid & 3;
    const int ntb = (wid >> 2) * NT;

    floatx8 acc[NT] = {};

#if HAVE_ASYNC_LDS
    auto issue = [&](int kc, _Float16* buf) {
        const _Float16* Wc = Wh + (size_t)kc * 32 * N;
        for (int u = tid; u < (32 * N) / 8; u += 256)
            __builtin_amdgcn_global_load_async_to_lds_b128(
                (as1_intx4*)(Wc + (size_t)u * 8),
                (as3_intx4*)(buf + (size_t)u * 8), 0, 0);
    };
    // Prologue: start streaming chunk 0 (previous stage's trailing barrier
    // guarantees both staging buffers are no longer being read).
    issue(0, ws0);
#pragma unroll 1
    for (int kc = 0; kc < KC; ++kc) {
        _Float16* cur = (kc & 1) ? ws1 : ws0;
        _Float16* nxt = (kc & 1) ? ws0 : ws1;
        wait_asynccnt0();   // my copies for chunk kc have landed
        __syncthreads();    // everyone's copies landed; prior reads retired
        if (kc + 1 < KC) issue(kc + 1, nxt); // overlap with compute below

        half16 a = load_a_frag(src, mtile * 16, src_ld, kc * 32);
#pragma unroll
        for (int t = 0; t < NT; ++t) {
            half16 bf = load_b_frag(cur, ntb + t);
            acc[t] = wmma_f16(a, bf, acc[t]);
        }
    }
#else
#pragma unroll 1
    for (int kc = 0; kc < KC; ++kc) {
        __syncthreads();
        const _Float16* Wc = Wh + (size_t)kc * 32 * N;
        if (kc + 1 < KC)
            __builtin_prefetch(Wh + (size_t)(kc + 1) * 32 * N, 0, 1);
        for (int u = tid; u < (32 * N) / 8; u += 256)
            *(half8*)(ws0 + (size_t)u * 8) = *(const half8*)(Wc + (size_t)u * 8);
        __syncthreads();

        half16 a = load_a_frag(src, mtile * 16, src_ld, kc * 32);
#pragma unroll
        for (int t = 0; t < NT; ++t) {
            half16 bf = load_b_frag(ws0, ntb + t);
            acc[t] = wmma_f16(a, bf, acc[t]);
        }
    }
#endif
    __syncthreads(); // all reads of src finished (src may alias dstL)

    // C layout: lane l, reg r -> m = r + 8*(l>>4), n = l&15 within tile.
#pragma unroll
    for (int t = 0; t < NT; ++t) {
        const int n0 = (ntb + t) * 16 + (lane & 15);
        const float bv = bias ? bias[n0] : 0.0f;
#pragma unroll
        for (int r = 0; r < 8; ++r) {
            const int m = mtile * 16 + r + 8 * (lane >> 4);
            float v = acc[t][r] + bv;
            if (relu) v = fmaxf(v, 0.0f);
            if (dstL) dstL[(size_t)m * N + n0] = (_Float16)v;
            if (dstG) dstG[(size_t)m * gld + n0] = v;
        }
    }
    __syncthreads();
}

// Row-wise softmax over a 64x64 f16 LDS tile (one thread per row).
__device__ inline void softmax64(_Float16* buf) {
    const int tid = threadIdx.x;
    if (tid < 64) {
        float mx = -3.0e38f;
        for (int k = 0; k < 64; ++k) mx = fmaxf(mx, (float)buf[tid * 64 + k]);
        float s = 0.0f;
        for (int k = 0; k < 64; ++k) s += expf((float)buf[tid * 64 + k] - mx);
        const float inv = 1.0f / s;
        for (int k = 0; k < 64; ++k)
            buf[tid * 64 + k] = (_Float16)(expf((float)buf[tid * 64 + k] - mx) * inv);
    }
    __syncthreads();
}

// ---------------------------------------------------------------------------
// Kernel 1: shift MLP + softmax -> shift_part (B x 512), v_shift (B x 256).
// ---------------------------------------------------------------------------
__global__ __launch_bounds__(256) void k_shift_mlp(
    const float* __restrict__ shift_bits,
    const _Float16* __restrict__ sd_w1h, const float* __restrict__ sd_b1,
    const _Float16* __restrict__ sd_w2h, const float* __restrict__ sd_b2,
    const _Float16* __restrict__ sd_w3h, const float* __restrict__ sd_b3,
    const _Float16* __restrict__ ix_w1bh, const _Float16* __restrict__ v_w1bh,
    float* __restrict__ shiftp, float* __restrict__ vshift) {
    extern __shared__ char smem[];
    _Float16* hbuf = (_Float16*)smem;    // 64 x 512
    _Float16* ws0  = hbuf + 64 * 512;    // 32 x 512
    _Float16* ws1  = ws0 + 32 * 512;     // 32 x 512
    _Float16* inbuf = ws1 + 32 * 512;    // 64 x 64
    _Float16* lgts  = inbuf + 64 * 64;   // 64 x 64

    const int b0 = blockIdx.x * 64;
    const int tid = threadIdx.x;

    for (int e4 = tid; e4 < (64 * 64) / 4; e4 += 256) {
        floatx4 v = *(const floatx4*)(shift_bits + (size_t)b0 * 64 + e4 * 4);
        half4 r;
#pragma unroll
        for (int q = 0; q < 4; ++q) r[q] = (_Float16)v[q];
        *(half4*)(inbuf + e4 * 4) = r;
    }
    __syncthreads();

    gemm64<512, 64 >(inbuf, 64, sd_w1h, sd_b1, ws0, ws1, hbuf, nullptr, 0, true);
    gemm64<512, 512>(hbuf, 512, sd_w2h, sd_b2, ws0, ws1, hbuf, nullptr, 0, true);
    gemm64<64,  512>(hbuf, 512, sd_w3h, sd_b3, ws0, ws1, lgts, nullptr, 0, false);
    softmax64(lgts); // lgts := shift_soft (f16)

    gemm64<512, 64>(lgts, 64, ix_w1bh, nullptr, ws0, ws1, nullptr,
                    shiftp + (size_t)b0 * 512, 512, false);
    gemm64<256, 64>(lgts, 64, v_w1bh, nullptr, ws0, ws1, nullptr,
                    vshift + (size_t)b0 * 256, 256, false);
}

// ---------------------------------------------------------------------------
// Kernel 2: per-batch fused pointer network (one WG per batch row).
// ---------------------------------------------------------------------------
__global__ __launch_bounds__(256) void k_pointer(
    const float* __restrict__ a_bits,
    const float* __restrict__ ix_w1, const float* __restrict__ ix_b1,
    const _Float16* __restrict__ ix_w2h, const float* __restrict__ ix_b2,
    const _Float16* __restrict__ ix_w3h, const float* __restrict__ ix_b3,
    const float* __restrict__ v_w1, const float* __restrict__ v_b1,
    const float* __restrict__ v_w2, const float* __restrict__ v_b2,
    const float* __restrict__ shiftp, const float* __restrict__ vshift,
    float* __restrict__ out) {
    extern __shared__ char smem[];
    _Float16* hbuf = (_Float16*)smem;  // 64 x 512 (h1, then h2 in place)
    _Float16* ws0  = hbuf + 64 * 512;  // 32 x 512
    _Float16* ws1  = ws0 + 32 * 512;   // 32 x 512
    _Float16* lgts = ws1 + 32 * 512;   // 64 x 64

    const int b = blockIdx.x;
    const int tid = threadIdx.x;

    // h1[i,n] = relu(ix_w1[i,n] + shift_part[b,n] + ix_b1[n]) -> f16 LDS
    for (int e4 = tid; e4 < (64 * 512) / 4; e4 += 256) {
        const int i = (e4 * 4) >> 9, n = (e4 * 4) & 511;
        floatx4 w = *(const floatx4*)(ix_w1 + (size_t)i * 512 + n);
        floatx4 s = *(const floatx4*)(shiftp + (size_t)b * 512 + n);
        floatx4 bb = *(const floatx4*)(ix_b1 + n);
        half4 r;
#pragma unroll
        for (int q = 0; q < 4; ++q)
            r[q] = (_Float16)fmaxf(w[q] + s[q] + bb[q], 0.0f);
        *(half4*)(hbuf + e4 * 4) = r;
    }
    __syncthreads();

    gemm64<512, 512>(hbuf, 512, ix_w2h, ix_b2, ws0, ws1, hbuf, nullptr, 0, true);
    gemm64<64,  512>(hbuf, 512, ix_w3h, ix_b3, ws0, ws1, lgts, nullptr, 0, false);
    softmax64(lgts); // lgts := idx_soft (f16)

    if (tid < 64) {
        const int i = tid;
        float pointed = 0.0f;
        for (int k = 0; k < 64; ++k)
            pointed += (float)lgts[i * 64 + k] * a_bits[(size_t)b * 64 + k];

        float va = 0.0f;
        for (int j = 0; j < 256; ++j) {
            const float vh = fmaxf(v_w1[(size_t)i * 256 + j] +
                                   vshift[(size_t)b * 256 + j] + v_b1[j], 0.0f);
            va += vh * v_w2[j];
        }
        const float lg = va + v_b2[0];
        out[(size_t)b * 64 + i] = pointed * (1.0f / (1.0f + expf(-lg)));
    }
}

// ---------------------------------------------------------------------------
extern "C" void kernel_launch(void* const* d_in, const int* in_sizes, int n_in,
                              void* d_out, int out_size, void* d_ws, size_t ws_size,
                              hipStream_t stream) {
    (void)in_sizes; (void)n_in; (void)out_size; (void)ws_size;
    const float* a_bits     = (const float*)d_in[0];
    const float* shift_bits = (const float*)d_in[1];
    const float* sd_w1 = (const float*)d_in[2];
    const float* sd_b1 = (const float*)d_in[3];
    const float* sd_w2 = (const float*)d_in[4];
    const float* sd_b2 = (const float*)d_in[5];
    const float* sd_w3 = (const float*)d_in[6];
    const float* sd_b3 = (const float*)d_in[7];
    const float* ix_w1 = (const float*)d_in[8];
    const float* ix_b1 = (const float*)d_in[9];
    const float* ix_w2 = (const float*)d_in[10];
    const float* ix_b2 = (const float*)d_in[11];
    const float* ix_w3 = (const float*)d_in[12];
    const float* ix_b3 = (const float*)d_in[13];
    const float* v_w1  = (const float*)d_in[14];
    const float* v_b1  = (const float*)d_in[15];
    const float* v_w2  = (const float*)d_in[16];
    const float* v_b2  = (const float*)d_in[17];

    // Workspace layout: f32 intermediates, then swizzled f16 weights.
    float* shiftp = (float*)d_ws;                  // 2048 x 512
    float* vshift = shiftp + (size_t)BATCH * 512;  // 2048 x 256
    _Float16* wpool = (_Float16*)(vshift + (size_t)BATCH * 256);
    _Float16* sd_w1h  = wpool;                  // 64*512
    _Float16* sd_w2h  = sd_w1h + 64 * 512;      // 512*512
    _Float16* sd_w3h  = sd_w2h + 512 * 512;     // 512*64
    _Float16* ix_w1bh = sd_w3h + 512 * 64;      // 64*512
    _Float16* v_w1bh  = ix_w1bh + 64 * 512;     // 64*256
    _Float16* ix_w2h  = v_w1bh + 64 * 256;      // 512*512
    _Float16* ix_w3h  = ix_w2h + 512 * 512;     // 512*64

    auto swz = [&](const float* W, _Float16* O, int K, int N) {
        const int total = K * N;
        k_swizzle<<<(total + 255) / 256, 256, 0, stream>>>(W, O, K, N);
    };
    swz(sd_w1, sd_w1h, 64, 512);
    swz(sd_w2, sd_w2h, 512, 512);
    swz(sd_w3, sd_w3h, 512, 64);
    swz(ix_w1 + (size_t)64 * 512, ix_w1bh, 64, 512);
    swz(v_w1 + (size_t)64 * 256, v_w1bh, 64, 256);
    swz(ix_w2, ix_w2h, 512, 512);
    swz(ix_w3, ix_w3h, 512, 64);

    const size_t sh1 = (size_t)(64 * 512 + 2 * 32 * 512 + 64 * 64 + 64 * 64) * 2;
    const size_t sh2 = (size_t)(64 * 512 + 2 * 32 * 512 + 64 * 64) * 2;

    k_shift_mlp<<<BATCH / 64, 256, sh1, stream>>>(
        shift_bits, sd_w1h, sd_b1, sd_w2h, sd_b2, sd_w3h, sd_b3,
        ix_w1bh, v_w1bh, shiftp, vshift);

    k_pointer<<<BATCH, 256, sh2, stream>>>(
        a_bits, ix_w1, ix_b1, ix_w2h, ix_b2, ix_w3h, ix_b3,
        v_w1, v_b1, v_w2, v_b2, shiftp, vshift, (float*)d_out);
}
